// FastRCNNHead_65558380806388
// MI455X (gfx1250) — matrix operand order repaired and checked by reading, not compiled, not run on hardware
//
#include <hip/hip_runtime.h>

typedef __attribute__((ext_vector_type(16))) __bf16 v16bf;
typedef __attribute__((ext_vector_type(8)))  float  v8f;

// ---------------- problem constants ----------------
#define NB        2
#define NPER      512
#define MROIS     1024        // NB*NPER
#define CCH       256
#define HH        50
#define WW        50
#define POOLP     7
#define FDIM      12544       // CCH*49
#define HID       1024
#define NCLS      81
#define NREG      324
#define NHEAD     405         // NCLS + NREG
#define NHPAD     448         // 28 tiles -> divisible by 4-tile wave strip
#define KT1       (FDIM/32)   // 392
#define KT2       (HID/32)    // 32
#define MT        (MROIS/16)  // 64
#define NT_HID    (HID/16)    // 64
#define NT_HEAD   (NHPAD/16)  // 28
#define KEEPN     100
#define CONF_THR  0.05f
#define NMS_THR   0.5f
#define MIN_SIZE  0.001f

// ---------------- helpers ----------------
__device__ __forceinline__ unsigned short f2bf(float f) {
    unsigned int u = __float_as_uint(f);
    u += 0x7fffu + ((u >> 16) & 1u);          // round-to-nearest-even
    return (unsigned short)(u >> 16);
}

// u16 index of element (m,k) inside packed WMMA A-layout (16x32 bf16 tiles)
__device__ __forceinline__ int apos(int m, int k, int Kt) {
    int mt = m >> 4, kt = k >> 5, kl = k & 31;
    int hi = kl >> 4;                 // K>=16 half -> vgpr 4..7
    int rem = kl & 15;
    int lane = ((rem >> 3) << 4) + (m & 15);
    int v = hi * 4 + ((rem & 7) >> 1);
    return (((mt * Kt + kt) * 32 + lane) << 4) + (v << 1) + (kl & 1);
}

// u16 index of element (k,n) inside packed WMMA B-layout (32x16 bf16 tiles)
__device__ __forceinline__ int bpos(int k, int n, int Nt) {
    int kt = k >> 5, kl = k & 31;
    int hi = kl >> 4;                 // K>=16 half -> vgpr 4..7
    int rem = kl & 15;
    int lane = ((rem >> 3) << 4) + (n & 15);
    int nt = n >> 4;
    int j = hi * 4 + ((rem & 7) >> 1);
    return (((kt * Nt + nt) * 32 + lane) << 4) + (j << 1) + (kl & 1);
}

// ---------------- pack kernels ----------------
__global__ void packB_kernel(const float* __restrict__ src, unsigned short* __restrict__ dst,
                             int K, int N, int Nt) {
    long long i = (long long)blockIdx.x * blockDim.x + threadIdx.x;
    if (i >= (long long)K * N) return;
    int k = (int)(i / N), n = (int)(i % N);
    dst[bpos(k, n, Nt)] = f2bf(src[i]);
}

__global__ void packBhead_kernel(const float* __restrict__ Wc, const float* __restrict__ Wr,
                                 unsigned short* __restrict__ dst) {
    long long i = (long long)blockIdx.x * blockDim.x + threadIdx.x;
    if (i >= (long long)HID * NHPAD) return;
    int k = (int)(i / NHPAD), n = (int)(i % NHPAD);
    float v = 0.f;
    if (n < NCLS)       v = Wc[(long long)k * NCLS + n];
    else if (n < NHEAD) v = Wr[(long long)k * NREG + (n - NCLS)];
    dst[bpos(k, n, NT_HEAD)] = f2bf(v);
}

__global__ void headbias_kernel(const float* __restrict__ bc, const float* __restrict__ br,
                                float* __restrict__ bh) {
    int n = blockIdx.x * blockDim.x + threadIdx.x;
    if (n >= NHPAD) return;
    bh[n] = (n < NCLS) ? bc[n] : ((n < NHEAD) ? br[n - NCLS] : 0.f);
}

__global__ void packA_kernel(const float* __restrict__ src, unsigned short* __restrict__ dst,
                             int Mr, int K, int Kt) {
    long long i = (long long)blockIdx.x * blockDim.x + threadIdx.x;
    if (i >= (long long)Mr * K) return;
    int m = (int)(i / K), k = (int)(i % K);
    dst[apos(m, k, Kt)] = f2bf(src[i]);
}

// ---------------- RoI max pool -> packed bf16 A ----------------
__global__ void __launch_bounds__(256)
roipool_kernel(const float* __restrict__ fmaps, const float* __restrict__ rois,
               unsigned short* __restrict__ featsPack) {
    int roi = blockIdx.x;              // 0..1023
    int img = roi >> 9;                // /512
    const float* rp = rois + (long long)roi * 4;
    int x1 = (int)rintf(rp[0]), y1 = (int)rintf(rp[1]);
    int x2 = (int)rintf(rp[2]), y2 = (int)rintf(rp[3]);
    float roiw = (float)max(x2 - x1 + 1, 1);
    float roih = (float)max(y2 - y1 + 1, 1);
    float binh = roih / (float)POOLP, binw = roiw / (float)POOLP;
    int c = threadIdx.x;               // 256 threads == 256 channels
    const float* fp = fmaps + ((long long)img * CCH + c) * (HH * WW);
    for (int ph = 0; ph < POOLP; ++ph) {
        int hs = min(max((int)floorf(ph * binh) + y1, 0), HH);
        int he = min(max((int)ceilf((ph + 1) * binh) + y1, 0), HH);
        for (int pw = 0; pw < POOLP; ++pw) {
            int wsb = min(max((int)floorf(pw * binw) + x1, 0), WW);
            int web = min(max((int)ceilf((pw + 1) * binw) + x1, 0), WW);
            float mx = -1e30f;
            for (int h = hs; h < he; ++h) {
                const float* row = fp + h * WW;
                for (int w = wsb; w < web; ++w) mx = fmaxf(mx, row[w]);
            }
            float v = (he <= hs || web <= wsb) ? 0.f : mx;
            int k = c * 49 + ph * POOLP + pw;
            featsPack[apos(roi, k, KT1)] = f2bf(v);
        }
    }
}

// ---------------- bf16 WMMA GEMM: out = relu?(Apack @ Bpack + bias) ----------------
// Requires Nt % 4 == 0 and Mt % 2 == 0.
// Each wave computes a 32x64 strip: 2 A-tiles x 4 B-tiles = 8 WMMA accumulators.
__global__ void __launch_bounds__(256)
gemm_wmma_bf16(const __bf16* __restrict__ Apack, const __bf16* __restrict__ Bpack,
               const float* __restrict__ bias, float* __restrict__ out,
               int Mt, int Nt, int Kt, int Npad, int relu) {
    int lane = threadIdx.x & 31;
    int wave = threadIdx.x >> 5;
    int ngroups = Nt >> 2;
    int mgroups = Mt >> 1;
    int job = blockIdx.x * 8 + wave;
    if (job >= mgroups * ngroups) return;     // wave-uniform: EXEC all ones inside
    int mt0 = (job / ngroups) * 2;
    int nt0 = (job % ngroups) * 4;

    float bv0 = bias[(nt0 + 0) * 16 + (lane & 15)];
    float bv1 = bias[(nt0 + 1) * 16 + (lane & 15)];
    float bv2 = bias[(nt0 + 2) * 16 + (lane & 15)];
    float bv3 = bias[(nt0 + 3) * 16 + (lane & 15)];
    v8f acc00, acc01, acc02, acc03;   // mtile mt0
    v8f acc10, acc11, acc12, acc13;   // mtile mt0+1
#pragma unroll
    for (int r = 0; r < 8; ++r) {
        acc00[r] = bv0; acc01[r] = bv1; acc02[r] = bv2; acc03[r] = bv3;
        acc10[r] = bv0; acc11[r] = bv1; acc12[r] = bv2; acc13[r] = bv3;
    }

    const v16bf* Ap0 = (const v16bf*)Apack + ((long long)(mt0 + 0) * Kt * 32 + lane);
    const v16bf* Ap1 = (const v16bf*)Apack + ((long long)(mt0 + 1) * Kt * 32 + lane);
    const v16bf* Bp  = (const v16bf*)Bpack + ((long long)nt0 * 32 + lane);
    const long long bStep = (long long)Nt * 32;

    for (int kt = 0; kt < Kt; ++kt) {
        v16bf a0 = Ap0[0];
        v16bf a1 = Ap1[0];
        __builtin_prefetch((const void*)(Ap0 + 32), 0, 3);
        __builtin_prefetch((const void*)(Ap1 + 32), 0, 3);
        v16bf b0 = Bp[0];
        v16bf b1 = Bp[32];
        v16bf b2 = Bp[64];
        v16bf b3 = Bp[96];
        acc00 = __builtin_amdgcn_wmma_f32_16x16x32_bf16(false, a0, false, b0, (short)0, acc00, false, false);
        acc10 = __builtin_amdgcn_wmma_f32_16x16x32_bf16(false, a1, false, b0, (short)0, acc10, false, false);
        acc01 = __builtin_amdgcn_wmma_f32_16x16x32_bf16(false, a0, false, b1, (short)0, acc01, false, false);
        acc11 = __builtin_amdgcn_wmma_f32_16x16x32_bf16(false, a1, false, b1, (short)0, acc11, false, false);
        acc02 = __builtin_amdgcn_wmma_f32_16x16x32_bf16(false, a0, false, b2, (short)0, acc02, false, false);
        acc12 = __builtin_amdgcn_wmma_f32_16x16x32_bf16(false, a1, false, b2, (short)0, acc12, false, false);
        acc03 = __builtin_amdgcn_wmma_f32_16x16x32_bf16(false, a0, false, b3, (short)0, acc03, false, false);
        acc13 = __builtin_amdgcn_wmma_f32_16x16x32_bf16(false, a1, false, b3, (short)0, acc13, false, false);
        Ap0 += 32;
        Ap1 += 32;
        Bp  += bStep;
    }

    int ncol = (lane & 15);
    int msub = ((lane >> 4) << 3);             // D layout: vgpr r -> M = r + (lane/16)*8
    {
        float* o0 = out + (long long)(mt0 * 16 + msub) * Npad + nt0 * 16 + ncol;
#pragma unroll
        for (int r = 0; r < 8; ++r) {
            float v0 = acc00[r], v1 = acc01[r], v2 = acc02[r], v3 = acc03[r];
            if (relu) {
                v0 = fmaxf(v0, 0.f); v1 = fmaxf(v1, 0.f);
                v2 = fmaxf(v2, 0.f); v3 = fmaxf(v3, 0.f);
            }
            float* orow = o0 + (long long)r * Npad;
            orow[0] = v0; orow[16] = v1; orow[32] = v2; orow[48] = v3;
        }
    }
    {
        float* o1 = out + (long long)((mt0 + 1) * 16 + msub) * Npad + nt0 * 16 + ncol;
#pragma unroll
        for (int r = 0; r < 8; ++r) {
            float v0 = acc10[r], v1 = acc11[r], v2 = acc12[r], v3 = acc13[r];
            if (relu) {
                v0 = fmaxf(v0, 0.f); v1 = fmaxf(v1, 0.f);
                v2 = fmaxf(v2, 0.f); v3 = fmaxf(v3, 0.f);
            }
            float* orow = o1 + (long long)r * Npad;
            orow[0] = v0; orow[16] = v1; orow[32] = v2; orow[48] = v3;
        }
    }
}

// ---------------- softmax-max / argmax / box decode ----------------
__global__ void scorebox_kernel(const float* __restrict__ out3, const float* __restrict__ rois,
                                float* __restrict__ boxes, float* __restrict__ scores,
                                int* __restrict__ preds, int* __restrict__ valid) {
    int roi = blockIdx.x * blockDim.x + threadIdx.x;
    if (roi >= MROIS) return;
    const float* lg = out3 + (long long)roi * NHPAD;
    float lmax = lg[0]; int pred = 0;
    for (int c = 1; c < NCLS; ++c) { float v = lg[c]; if (v > lmax) { lmax = v; pred = c; } }
    float s = 0.f;
    for (int c = 0; c < NCLS; ++c) s += expf(lg[c] - lmax);
    float score = 1.f / s;                      // exp(lmax-lmax)/sum
    float dx = lg[NCLS + pred * 4 + 0], dy = lg[NCLS + pred * 4 + 1];
    float dw = lg[NCLS + pred * 4 + 2], dh = lg[NCLS + pred * 4 + 3];
    const float* rp = rois + (long long)roi * 4;
    float w = rp[2] - rp[0], h = rp[3] - rp[1];
    float cx = rp[0] + 0.5f * w, cy = rp[1] + 0.5f * h;
    float pcx = dx * w + cx, pcy = dy * h + cy;
    float pw = expf(dw) * w, ph = expf(dh) * h;
    float bx1 = pcx - 0.5f * pw, by1 = pcy - 0.5f * ph;
    float bx2 = pcx + 0.5f * pw, by2 = pcy + 0.5f * ph;
    boxes[roi * 4 + 0] = bx1; boxes[roi * 4 + 1] = by1;
    boxes[roi * 4 + 2] = bx2; boxes[roi * 4 + 3] = by2;
    scores[roi] = score; preds[roi] = pred;
    valid[roi] = (pred != 0) && (score >= CONF_THR) &&
                 (bx2 - bx1 >= MIN_SIZE) && (by2 - by1 >= MIN_SIZE);
}

// ---------------- per-image sort + greedy NMS + top-100 ----------------
__global__ void __launch_bounds__(256)
nms_kernel(const float* __restrict__ boxes, const float* __restrict__ scores,
           const int* __restrict__ preds, const int* __restrict__ valid,
           float* __restrict__ dout) {
    __shared__ float skey[NPER]; __shared__ int sidx[NPER];
    __shared__ float sx1[NPER], sy1[NPER], sx2[NPER], sy2[NPER], sar[NPER], scl[NPER];
    __shared__ int   skeep[NPER];
    __shared__ float sred[256];
    int img = blockIdx.x, t = threadIdx.x;
    int base = img * NPER;

    // max |box| over all boxes of this image
    float mloc = 0.f;
    for (int e = 0; e < 2; ++e) {
        int i = t + e * 256;
        const float* b = boxes + (long long)(base + i) * 4;
        mloc = fmaxf(mloc, fmaxf(fmaxf(fabsf(b[0]), fabsf(b[1])),
                                 fmaxf(fabsf(b[2]), fabsf(b[3]))));
    }
    sred[t] = mloc; __syncthreads();
    for (int s = 128; s > 0; s >>= 1) {
        if (t < s) sred[t] = fmaxf(sred[t], sred[t + s]);
        __syncthreads();
    }
    float offscale = sred[0] + 1.f;

    // keys: valid ? score : -1 ; stable descending via (key desc, idx asc)
    for (int e = 0; e < 2; ++e) {
        int i = t + e * 256;
        skey[i] = valid[base + i] ? scores[base + i] : -1.f;
        sidx[i] = i;
    }
    __syncthreads();
    for (int k = 2; k <= NPER; k <<= 1) {
        for (int j = k >> 1; j > 0; j >>= 1) {
            for (int e = 0; e < 2; ++e) {
                int i = t + e * 256;
                int ixj = i ^ j;
                if (ixj > i) {
                    float ka = skey[i], kb = skey[ixj];
                    int ia = sidx[i], ib = sidx[ixj];
                    bool abf = (ka > kb) || (ka == kb && ia < ib);
                    bool desc = ((i & k) == 0);
                    if (desc != abf) {
                        skey[i] = kb; skey[ixj] = ka;
                        sidx[i] = ib; sidx[ixj] = ia;
                    }
                }
            }
            __syncthreads();
        }
    }

    // gather sorted boxes, add per-class offset (classes never overlap)
    for (int e = 0; e < 2; ++e) {
        int r = t + e * 256;
        int g = base + sidx[r];
        float c = (float)preds[g];
        float ofv = offscale * c;
        float x1 = boxes[g * 4 + 0] + ofv, y1 = boxes[g * 4 + 1] + ofv;
        float x2 = boxes[g * 4 + 2] + ofv, y2 = boxes[g * 4 + 3] + ofv;
        sx1[r] = x1; sy1[r] = y1; sx2[r] = x2; sy2[r] = y2;
        sar[r] = (x2 - x1) * (y2 - y1);
        scl[r] = c;
        skeep[r] = valid[g];
    }
    __syncthreads();

    // greedy suppression
    for (int i = 0; i < NPER; ++i) {
        bool alive = (skeep[i] != 0);
        if (alive) {
            float ax1 = sx1[i], ay1 = sy1[i], ax2 = sx2[i], ay2 = sy2[i], aar = sar[i];
            for (int e = 0; e < 2; ++e) {
                int j = t + e * 256;
                if (j > i && skeep[j]) {
                    float lx = fmaxf(ax1, sx1[j]), ly = fmaxf(ay1, sy1[j]);
                    float rx = fminf(ax2, sx2[j]), ry = fminf(ay2, sy2[j]);
                    float iw = fmaxf(rx - lx, 0.f), ih = fmaxf(ry - ly, 0.f);
                    float inter = iw * ih;
                    float iou = inter / (aar + sar[j] - inter + 1e-12f);
                    if (iou > NMS_THR) skeep[j] = 0;
                }
            }
        }
        __syncthreads();
    }

    // emit top-100 kept (already in descending-score order)
    if (t == 0) {
        float* o = dout + (long long)img * KEEPN * 6;
        int cnt = 0;
        for (int r = 0; r < NPER && cnt < KEEPN; ++r) {
            if (skeep[r]) {
                int g = base + sidx[r];
                o[cnt * 6 + 0] = boxes[g * 4 + 0];
                o[cnt * 6 + 1] = boxes[g * 4 + 1];
                o[cnt * 6 + 2] = boxes[g * 4 + 2];
                o[cnt * 6 + 3] = boxes[g * 4 + 3];
                o[cnt * 6 + 4] = skey[r];
                o[cnt * 6 + 5] = scl[r];
                ++cnt;
            }
        }
        for (; cnt < KEEPN; ++cnt)
            for (int q = 0; q < 6; ++q) o[cnt * 6 + q] = 0.f;
    }
}

// ---------------- host launcher ----------------
extern "C" void kernel_launch(void* const* d_in, const int* in_sizes, int n_in,
                              void* d_out, int out_size, void* d_ws, size_t ws_size,
                              hipStream_t stream) {
    const float* fmaps = (const float*)d_in[0];
    const float* rois  = (const float*)d_in[1];
    const float* W1    = (const float*)d_in[2];
    const float* b1    = (const float*)d_in[3];
    const float* W2    = (const float*)d_in[4];
    const float* b2    = (const float*)d_in[5];
    const float* Wc    = (const float*)d_in[6];
    const float* bc    = (const float*)d_in[7];
    const float* Wr    = (const float*)d_in[8];
    const float* br    = (const float*)d_in[9];
    float* out = (float*)d_out;

    char* ws = (char*)d_ws;
    size_t off = 0;
    auto take = [&](size_t bytes) -> void* {
        off = (off + 255) & ~(size_t)255;
        void* p = ws + off;
        off += bytes;
        return p;
    };
    unsigned short* featsPack = (unsigned short*)take((size_t)MROIS * FDIM * 2 + 1024);  // +prefetch slack
    unsigned short* w1Pack    = (unsigned short*)take((size_t)FDIM * HID * 2);
    unsigned short* w2Pack    = (unsigned short*)take((size_t)HID * HID * 2);
    unsigned short* whPack    = (unsigned short*)take((size_t)HID * NHPAD * 2);
    float* bh    = (float*)take((size_t)NHPAD * 4);
    float* hbuf  = (float*)take((size_t)MROIS * HID * 4);                          // h1 then h2
    unsigned short* hPack = (unsigned short*)take((size_t)MROIS * HID * 2 + 1024); // h1pack then h2pack
    float* out3  = (float*)take((size_t)MROIS * NHPAD * 4);
    float* boxes = (float*)take((size_t)MROIS * 4 * 4);
    float* scoresB = (float*)take((size_t)MROIS * 4);
    int* predsB  = (int*)take((size_t)MROIS * 4);
    int* validB  = (int*)take((size_t)MROIS * 4);

    // pack weights into WMMA-B layouts
    {
        long long n1 = (long long)FDIM * HID;
        packB_kernel<<<(unsigned)((n1 + 255) / 256), 256, 0, stream>>>(W1, w1Pack, FDIM, HID, NT_HID);
        long long n2 = (long long)HID * HID;
        packB_kernel<<<(unsigned)((n2 + 255) / 256), 256, 0, stream>>>(W2, w2Pack, HID, HID, NT_HID);
        long long nh = (long long)HID * NHPAD;
        packBhead_kernel<<<(unsigned)((nh + 255) / 256), 256, 0, stream>>>(Wc, Wr, whPack);
        headbias_kernel<<<2, 256, 0, stream>>>(bc, br, bh);
    }

    // RoI max pool straight into WMMA-A layout (bf16)
    roipool_kernel<<<MROIS, 256, 0, stream>>>(fmaps, rois, featsPack);

    // GEMM1: [1024,12544]x[12544,1024]+b1, ReLU ; (64/2)*(64/4)=512 wave jobs -> 64 blocks
    gemm_wmma_bf16<<<64, 256, 0, stream>>>((const __bf16*)featsPack, (const __bf16*)w1Pack,
                                           b1, hbuf, MT, NT_HID, KT1, HID, 1);
    packA_kernel<<<(unsigned)(((long long)MROIS * HID + 255) / 256), 256, 0, stream>>>(
        hbuf, hPack, MROIS, HID, KT2);

    // GEMM2: [1024,1024]x[1024,1024]+b2, ReLU ; 512 jobs -> 64 blocks
    gemm_wmma_bf16<<<64, 256, 0, stream>>>((const __bf16*)hPack, (const __bf16*)w2Pack,
                                           b2, hbuf, MT, NT_HID, KT2, HID, 1);
    packA_kernel<<<(unsigned)(((long long)MROIS * HID + 255) / 256), 256, 0, stream>>>(
        hbuf, hPack, MROIS, HID, KT2);

    // GEMM3 (fused cls+reg heads): [1024,1024]x[1024,448]+bh ; (64/2)*(28/4)=224 jobs -> 28 blocks
    gemm_wmma_bf16<<<28, 256, 0, stream>>>((const __bf16*)hPack, (const __bf16*)whPack,
                                           bh, out3, MT, NT_HEAD, KT2, NHPAD, 0);

    scorebox_kernel<<<4, 256, 0, stream>>>(out3, rois, boxes, scoresB, predsB, validB);
    nms_kernel<<<NB, 256, 0, stream>>>(boxes, scoresB, predsB, validB, out);
}
